// PlainMamba2D_18391049962091
// MI455X (gfx1250) — matrix operand order repaired and checked by reading, not compiled, not run on hardware
//
#include <hip/hip_runtime.h>
#include <hip/hip_fp16.h>

#define D_MODEL 192
#define D_STATE 16
#define D_INNER 384
#define DT_RANK 12
#define BATCH 16
#define HH 48
#define WW 48
#define LL (HH*WW)          // 2304
#define EPSV 1e-5f

typedef _Float16 v16h __attribute__((ext_vector_type(16)));
typedef float    v8f  __attribute__((ext_vector_type(8)));

union FragU { v16h v; uint4 u[2]; };

// ---- Tensor Data Mover support (guarded: falls back to manual loads) -------
#if defined(__HIP_DEVICE_COMPILE__) && __has_builtin(__builtin_amdgcn_tensor_load_to_lds) && __has_builtin(__builtin_amdgcn_s_wait_tensorcnt)
#define USE_TDM 1
#else
#define USE_TDM 0
#endif

#if USE_TDM
typedef unsigned int u32x4 __attribute__((ext_vector_type(4)));
typedef int          i32x8 __attribute__((ext_vector_type(8)));
typedef int          i32x4 __attribute__((ext_vector_type(4)));

// DMA a 2D tile (rows x 48 fp32, row stride LL elements) from global -> LDS.
// D# built per CDNA5 ISA ch.8: group0 {count=1, lds_addr, global_addr, type=2},
// group1 {data_size=4B, tensor dims, tile dims, dim0 stride}; groups 2/3 zero.
__device__ __forceinline__ void tdm_load_2d(const float* gsrc, const void* ldsDst,
                                            int rows)
{
    unsigned long long ga = (unsigned long long)(const void*)gsrc;
    unsigned ldsOff = (unsigned)(unsigned long long)ldsDst;  // flat low 32b = LDS offset
    u32x4 g0;
    g0[0] = 1u;                                             // count=1 (valid)
    g0[1] = ldsOff;                                         // lds_addr
    g0[2] = (unsigned)(ga & 0xFFFFFFFFull);                 // global_addr[31:0]
    g0[3] = (unsigned)((ga >> 32) & 0x1FFFFFFull) | (2u << 30); // [56:32] | type=2
    i32x8 g1;
    g1[0] = (int)(2u << 16);                                // data_size = 4B
    g1[1] = (int)(((unsigned)LL & 0xFFFFu) << 16);          // tensor_dim0 = LL
    g1[2] = (int)((unsigned)rows << 16);                    // tensor_dim1 = rows
    g1[3] = (int)(48u << 16);                               // tile_dim0 = 48
    g1[4] = rows;                                           // tile_dim1 = rows
    g1[5] = LL;                                             // tensor_dim0_stride
    g1[6] = 0;
    g1[7] = 0;
    i32x4 z4 = {0, 0, 0, 0};
#if defined(__clang_major__) && (__clang_major__ >= 23)
    i32x8 z8 = {0, 0, 0, 0, 0, 0, 0, 0};
    __builtin_amdgcn_tensor_load_to_lds(g0, g1, z4, z4, z8, 0);
#else
    __builtin_amdgcn_tensor_load_to_lds(g0, g1, z4, z4, 0);
#endif
}
#endif // USE_TDM

// ---------------------------------------------------------------------------
// Batched GEMM  Out[b,m,n] = BN( sum_k A[m,k]*B[b,k,n] + bias[m] )
// WMMA f32<-f16 16x16x32.  Block tile 32(M) x 64(N): 4 waves, each wave owns
// one 16-col B fragment and runs TWO wmmas.
// ---------------------------------------------------------------------------
template<bool BN_CONTIG>
__global__ __launch_bounds__(128)
void gemm_bn_kernel(const float* __restrict__ A, int M, int K,
                    const float* __restrict__ Bmat, long batchStrideB,
                    long strideBK, long strideBN,
                    float* __restrict__ Out,
                    const float* __restrict__ gam, const float* __restrict__ bet,
                    const float* __restrict__ mean, const float* __restrict__ var,
                    const float* __restrict__ bias)
{
    __shared__ __align__(16) _Float16 lds_a[32*32];
    __shared__ __align__(16) _Float16 lds_b[64*32];
    const int tid = threadIdx.x;
    const int wave = tid >> 5;
    const int lane = tid & 31;
    const int half16 = lane >> 4;
    const int l16 = lane & 15;
    const int n0 = blockIdx.x * 64;
    const int m0 = blockIdx.y * 32;
    const int b  = blockIdx.z;
    const float* Bb = Bmat + (long)b * batchStrideB;

    v8f acc0 = {}, acc1 = {};
    for (int k0 = 0; k0 < K; k0 += 32) {
        #pragma unroll
        for (int p = 0; p < 8; ++p) {
            int idx = tid + p*128;
            int m = idx >> 5, k = idx & 31;
            lds_a[idx] = (_Float16)A[(m0+m)*K + (k0+k)];
        }
        if (BN_CONTIG) {
            #pragma unroll
            for (int p = 0; p < 16; ++p) {
                int idx = tid + p*128;
                int k = idx >> 6, n = idx & 63;
                lds_b[n*32 + k] = (_Float16)Bb[(long)(k0+k)*strideBK + (n0+n)];
            }
        } else {
            #pragma unroll
            for (int p = 0; p < 16; ++p) {
                int idx = tid + p*128;
                int n = idx >> 5, k = idx & 31;
                lds_b[n*32 + k] = (_Float16)Bb[(long)(n0+n)*strideBN + (k0+k)];
            }
        }
        __syncthreads();
        FragU af0, af1, bf;
        const uint4* pa0 = reinterpret_cast<const uint4*>(&lds_a[l16*32 + half16*8]);
        af0.u[0] = pa0[0];
        af0.u[1] = pa0[2];
        const uint4* pa1 = reinterpret_cast<const uint4*>(&lds_a[(16+l16)*32 + half16*8]);
        af1.u[0] = pa1[0];
        af1.u[1] = pa1[2];
        const int nloc = wave*16 + l16;
        const uint4* pb = reinterpret_cast<const uint4*>(&lds_b[nloc*32 + half16*8]);
        bf.u[0] = pb[0];
        bf.u[1] = pb[2];
        acc0 = __builtin_amdgcn_wmma_f32_16x16x32_f16(false, af0.v, false, bf.v,
                                                      (short)0, acc0, false, false);
        acc1 = __builtin_amdgcn_wmma_f32_16x16x32_f16(false, af1.v, false, bf.v,
                                                      (short)0, acc1, false, false);
        __syncthreads();
    }
    const int col = n0 + wave*16 + l16;
    #pragma unroll
    for (int r = 0; r < 8; ++r) {
        int m = m0 + r + half16*8;
        float s  = gam[m] * rsqrtf(var[m] + EPSV);
        float sh = (bias[m] - mean[m]) * s + bet[m];
        Out[((long)b*M + m)*LL + col] = acc0[r]*s + sh;
        int m2 = m + 16;
        float s2  = gam[m2] * rsqrtf(var[m2] + EPSV);
        float sh2 = (bias[m2] - mean[m2]) * s2 + bet[m2];
        Out[((long)b*M + m2)*LL + col] = acc1[r]*s2 + sh2;
    }
}

// ---------------------------------------------------------------------------
// Depthwise 7x7 conv + bias + SiLU.  h1,(B,E,L) -> xc,(B,E,L)
// ---------------------------------------------------------------------------
__global__ __launch_bounds__(256)
void dwconv_silu_kernel(const float* __restrict__ h1, const float* __restrict__ w_dw,
                        const float* __restrict__ b_dw, float* __restrict__ xc)
{
    unsigned idx = blockIdx.x * 256u + threadIdx.x;
    int l = idx % LL;
    unsigned be = idx / LL;
    int e = be % D_INNER;
    int hh = l / WW, ww = l % WW;
    const float* src = h1 + (long)be * LL;
    const float* wk  = w_dw + e*49;
    float acc = b_dw[e];
    #pragma unroll
    for (int dy = 0; dy < 7; ++dy) {
        int y = hh + dy - 3;
        if (y < 0 || y >= HH) continue;
        #pragma unroll
        for (int dx = 0; dx < 7; ++dx) {
            int x = ww + dx - 3;
            if (x < 0 || x >= WW) continue;
            acc += src[y*WW + x] * wk[dy*7 + dx];
        }
    }
    xc[idx] = acc / (1.f + __expf(-acc));             // SiLU
}

// ---------------------------------------------------------------------------
// x_dbl[b,j,l] = sum_e xc[b,e,l] * w_xproj[j,e]      (j = 0..43)
// ---------------------------------------------------------------------------
__global__ __launch_bounds__(256)
void xproj_kernel(const float* __restrict__ xc, const float* __restrict__ w_xproj,
                  float* __restrict__ xdbl)
{
    int j = blockIdx.x % 44;
    int b = blockIdx.x / 44;
    int l = blockIdx.y * 256 + threadIdx.x;
    const float* xb = xc + (long)b * D_INNER * LL + l;
    const float* wr = w_xproj + j * D_INNER;
    float acc = 0.f;
    #pragma unroll 4
    for (int e = 0; e < D_INNER; ++e)
        acc += xb[(long)e*LL] * wr[e];
    xdbl[(long)blockIdx.x * LL + l] = acc;
}

// ---------------------------------------------------------------------------
// delta[b,e,l] = softplus( dtr . w_dt[e] + 2*b_dt[e] )
// ---------------------------------------------------------------------------
__global__ __launch_bounds__(256)
void delta_kernel(const float* __restrict__ xdbl, const float* __restrict__ w_dt,
                  const float* __restrict__ b_dt, float* __restrict__ delta)
{
    int e = blockIdx.x % D_INNER;
    int b = blockIdx.x / D_INNER;
    int l = blockIdx.y * 256 + threadIdx.x;
    const float* dtr = xdbl + (long)b * 44 * LL + l;
    const float* wr = w_dt + e * DT_RANK;
    float acc = 2.f * b_dt[e];
    #pragma unroll
    for (int r = 0; r < DT_RANK; ++r)
        acc += dtr[(long)r*LL] * wr[r];
    delta[(long)blockIdx.x * LL + l] = (acc > 20.f) ? acc : log1pf(__expf(acc));
}

// ---------------------------------------------------------------------------
// Selective scan.  4 states per lane; quad shfl-xor reduction for y.
// Snake order closed-form.  Operand staging per 48-step row chunk is
// double-buffered: TDM (tensor_load_to_lds + s_wait_tensorcnt) fetches the
// next chunk while the serial recurrence runs; fallback path uses manual
// loads + global_prefetch_b8.
// ---------------------------------------------------------------------------
__global__ __launch_bounds__(256)
void scan_kernel(const float* __restrict__ delta, const float* __restrict__ xc,
                 const float* __restrict__ xdbl, const float* __restrict__ A_log,
                 const float* __restrict__ Dp, const float* __restrict__ dir_Bs,
                 float* __restrict__ ybuf)
{
    __shared__ float s_delta[2][64*48];
    __shared__ float s_u[2][64*48];
    __shared__ float s_bc[2][32*48];  // rows 0..15: Bm(n), 16..31: Cm(n)
    const int tid = threadIdx.x;
    const int e_loc = tid >> 2;       // 0..63
    const int ng = tid & 3;           // state group: n = ng*4 + j
    const int b  = blockIdx.y;
    const int e0 = blockIdx.x * 64;
    const int e  = e0 + e_loc;

    float Aj[4], dB0[4], dB1[4], dB2[4], dB4[4];
    #pragma unroll
    for (int j2 = 0; j2 < 4; ++j2) {
        int n = ng*4 + j2;
        Aj[j2]  = -__expf(A_log[e*D_STATE + n]);
        dB0[j2] = dir_Bs[0*D_STATE + n];
        dB1[j2] = dir_Bs[1*D_STATE + n];
        dB2[j2] = dir_Bs[2*D_STATE + n];
        dB4[j2] = dir_Bs[4*D_STATE + n];
    }
    const float Dpe = Dp[e];
    const float* dbase  = delta + ((long)b*D_INNER + e0)*LL;
    const float* ubase  = xc    + ((long)b*D_INNER + e0)*LL;
    const float* bcbase = xdbl  + ((long)b*44 + 12)*LL;

#if USE_TDM
    const bool isWave0 = (tid >> 5) == 0;   // wave-uniform scalar branch
    if (isWave0) {
        tdm_load_2d(dbase,  &s_delta[0][0], 64);
        tdm_load_2d(ubase,  &s_u[0][0],     64);
        tdm_load_2d(bcbase, &s_bc[0][0],    32);
    }
#endif

    float st[4] = {0.f, 0.f, 0.f, 0.f};
    for (int row = 0; row < HH; ++row) {
        const int t0 = row * WW;
        const int cur = row & 1;
#if USE_TDM
        if (isWave0) {
            if (row + 1 < HH) {
                const int nxt = (row + 1) & 1;
                const int t1 = t0 + WW;
                tdm_load_2d(dbase + t1,  &s_delta[nxt][0], 64);
                tdm_load_2d(ubase + t1,  &s_u[nxt][0],     64);
                tdm_load_2d(bcbase + t1, &s_bc[nxt][0],    32);
                __builtin_amdgcn_s_wait_tensorcnt((short)3); // current chunk done
            } else {
                __builtin_amdgcn_s_wait_tensorcnt((short)0);
            }
        }
#else
        #pragma unroll
        for (int p = 0; p < 12; ++p) {
            int idx = tid + p*256;                // 64*48 elements
            int ee = idx / 48, kk = idx % 48;
            s_delta[cur][idx] = dbase[(long)ee*LL + t0 + kk];
            s_u[cur][idx]     = ubase[(long)ee*LL + t0 + kk];
        }
        #pragma unroll
        for (int p = 0; p < 6; ++p) {
            int idx = tid + p*256;                // 32*48 elements
            int nn = idx / 48, kk = idx % 48;
            s_bc[cur][idx] = bcbase[(long)nn*LL + t0 + kk];
        }
        if (row + 1 < HH) {
            const int t1 = t0 + WW;
            int pidx = tid * 12;
            int ee = pidx / 48, kk = pidx % 48;
            __builtin_prefetch(&dbase[(long)ee*LL + t1 + kk], 0, 1);
            __builtin_prefetch(&ubase[(long)ee*LL + t1 + kk], 0, 1);
            int qidx = tid * 6;
            int nn = qidx / 48, k2 = qidx % 48;
            __builtin_prefetch(&bcbase[(long)nn*LL + t1 + k2], 0, 1);
        }
#endif
        __syncthreads();
        const int odd = row & 1;
        for (int kk = 0; kk < WW; ++kk) {
            float ds = s_delta[cur][e_loc*48 + kk];
            int jcol = odd ? (47 - kk) : kk;
            float us = s_u[cur][e_loc*48 + jcol];
            float du = ds * us;
            float yp = 0.f;
            #pragma unroll
            for (int j2 = 0; j2 < 4; ++j2) {
                int n = ng*4 + j2;
                float dirB = (kk == 0) ? ((row == 0) ? dB0[j2] : dB4[j2])
                                       : (odd ? dB2[j2] : dB1[j2]);
                float Bn = s_bc[cur][n*48 + kk] + dirB;
                float Cn = s_bc[cur][(16 + n)*48 + kk];
                st[j2] = st[j2]*__expf(ds*Aj[j2]) + du*Bn;
                yp += st[j2]*Cn;
            }
            yp += __shfl_xor(yp, 1, 32);
            yp += __shfl_xor(yp, 2, 32);
            if (ng == 0) {
                int r = t0 + jcol;                // raster = order[t]
                ybuf[((long)b*LL + r)*D_INNER + e] = 4.f*(yp + us*Dpe);
            }
        }
        __syncthreads();
    }
}

// ---------------------------------------------------------------------------
// LayerNorm over E + ReLU.  ybuf,(B,L,E) -> z,(B,L,E)
// ---------------------------------------------------------------------------
__global__ __launch_bounds__(128)
void ln_relu_kernel(const float* __restrict__ ybuf, const float* __restrict__ ln_g,
                    const float* __restrict__ ln_b, float* __restrict__ z)
{
    __shared__ float s_sum[4], s_sq[4];
    long bl = blockIdx.x;                         // b*LL + l
    const float* yr = ybuf + bl*D_INNER;
    int tid = threadIdx.x;
    float v0 = yr[tid], v1 = yr[tid+128], v2 = yr[tid+256];
    float s = v0+v1+v2;
    float q = v0*v0 + v1*v1 + v2*v2;
    #pragma unroll
    for (int m = 16; m >= 1; m >>= 1) {
        s += __shfl_xor(s, m, 32);
        q += __shfl_xor(q, m, 32);
    }
    int wave = tid >> 5;
    if ((tid & 31) == 0) { s_sum[wave] = s; s_sq[wave] = q; }
    __syncthreads();
    s = s_sum[0]+s_sum[1]+s_sum[2]+s_sum[3];
    q = s_sq[0]+s_sq[1]+s_sq[2]+s_sq[3];
    float mu  = s * (1.f/D_INNER);
    float var = q * (1.f/D_INNER) - mu*mu;
    float rs  = rsqrtf(var + EPSV);
    float* zr = z + bl*D_INNER;
    zr[tid]     = fmaxf((v0-mu)*rs*ln_g[tid]     + ln_b[tid],     0.f);
    zr[tid+128] = fmaxf((v1-mu)*rs*ln_g[tid+128] + ln_b[tid+128], 0.f);
    zr[tid+256] = fmaxf((v2-mu)*rs*ln_g[tid+256] + ln_b[tid+256], 0.f);
}

// ---------------------------------------------------------------------------
extern "C" void kernel_launch(void* const* d_in, const int* in_sizes, int n_in,
                              void* d_out, int out_size, void* d_ws, size_t ws_size,
                              hipStream_t stream)
{
    (void)in_sizes; (void)n_in; (void)out_size; (void)ws_size;
    const float* x      = (const float*)d_in[0];
    const float* w_in   = (const float*)d_in[1];
    const float* b_in   = (const float*)d_in[2];
    const float* bn1_g  = (const float*)d_in[3];
    const float* bn1_b  = (const float*)d_in[4];
    const float* bn1_m  = (const float*)d_in[5];
    const float* bn1_v  = (const float*)d_in[6];
    const float* w_dw   = (const float*)d_in[7];
    const float* b_dw   = (const float*)d_in[8];
    const float* w_xproj= (const float*)d_in[9];
    const float* w_dt   = (const float*)d_in[10];
    const float* b_dt   = (const float*)d_in[11];
    const float* A_log  = (const float*)d_in[12];
    const float* Dp     = (const float*)d_in[13];
    const float* dir_Bs = (const float*)d_in[14];
    const float* ln_g   = (const float*)d_in[15];
    const float* ln_b   = (const float*)d_in[16];
    const float* w_out  = (const float*)d_in[17];
    const float* b_out  = (const float*)d_in[18];
    const float* bn2_g  = (const float*)d_in[19];
    const float* bn2_b  = (const float*)d_in[20];
    const float* bn2_m  = (const float*)d_in[21];
    const float* bn2_v  = (const float*)d_in[22];
    float* out = (float*)d_out;

    float* ws = (float*)d_ws;
    const long SZ_BEL = (long)BATCH * D_INNER * LL;   // 14,155,776 floats
    float* h1    = ws;                                 // (B,E,L)
    float* xc    = ws + SZ_BEL;                        // (B,E,L)
    float* xdbl  = ws + 2*SZ_BEL;                      // (B,44,L)
    float* delta = xdbl + (long)BATCH*44*LL;           // (B,E,L)
    float* ybuf  = delta + SZ_BEL;                     // (B,L,E)
    float* z     = h1;                                 // reuse dead h1 as (B,L,E)

    // 1) in-projection + BN1 (WMMA)
    gemm_bn_kernel<true><<<dim3(LL/64, D_INNER/32, BATCH), 128, 0, stream>>>(
        w_in, D_INNER, D_MODEL, x, (long)D_MODEL*LL, (long)LL, 1L,
        h1, bn1_g, bn1_b, bn1_m, bn1_v, b_in);

    // 2) depthwise 7x7 conv + SiLU
    dwconv_silu_kernel<<<(int)(SZ_BEL/256), 256, 0, stream>>>(h1, w_dw, b_dw, xc);

    // 3) x_dbl = x_conv @ w_xproj^T  (stored (B,44,L))
    xproj_kernel<<<dim3(BATCH*44, LL/256), 256, 0, stream>>>(xc, w_xproj, xdbl);

    // 4) delta = softplus(dtr @ w_dt^T + 2*b_dt)  (stored (B,E,L))
    delta_kernel<<<dim3(BATCH*D_INNER, LL/256), 256, 0, stream>>>(xdbl, w_dt, b_dt, delta);

    // 5) snake selective scan -> ybuf (B,L,E)
    scan_kernel<<<dim3(D_INNER/64, BATCH), 256, 0, stream>>>(
        delta, xc, xdbl, A_log, Dp, dir_Bs, ybuf);

    // 6) LayerNorm + ReLU -> z (B,L,E)
    ln_relu_kernel<<<BATCH*LL, 128, 0, stream>>>(ybuf, ln_g, ln_b, z);

    // 7) out-projection + BN2 (WMMA)
    gemm_bn_kernel<false><<<dim3(LL/64, D_MODEL/32, BATCH), 128, 0, stream>>>(
        w_out, D_MODEL, D_INNER, z, (long)LL*D_INNER, 1L, (long)D_INNER,
        out, bn2_g, bn2_b, bn2_m, bn2_v, b_out);
}
